// NeuralODE_SAC_ManifoldLoss_52467320488100
// MI455X (gfx1250) — matrix operand order repaired and checked by reading, not compiled, not run on hardware
//
#include <hip/hip_runtime.h>

// ---------------------------------------------------------------------------
// NeuralODE + SAC actor fused Euler integrator for gfx1250 (MI455X).
// One workgroup (256 threads = 8 wave32) owns 64 batch rows and runs all 63
// Euler steps with every GEMM on v_wmma_f32_16x16x32_f16. All weights live in
// LDS (fp16, transposed) for the whole kernel: 298.75 KB of the WGP's 320 KB.
// ---------------------------------------------------------------------------

typedef __attribute__((ext_vector_type(16))) _Float16 v16h;
typedef __attribute__((ext_vector_type(8)))  _Float16 v8h;
typedef __attribute__((ext_vector_type(8)))  float    v8f;

#define DEV __device__ __forceinline__

constexpr int Bq = 16384;   // batch
constexpr int Dq = 64;      // state dim
constexpr int Hq = 256;     // ODE hidden
constexpr int Tq = 64;      // time steps
constexpr int NTHREADS = 256;
constexpr int ROWS_PER_WG = 64;

// ------------------------- LDS layout (bytes) ------------------------------
constexpr size_t OFF_W1T  = 0;                         // half [256][64]
constexpr size_t OFF_W2T  = OFF_W1T + 256 * 64 * 2;    // half [256][256]
constexpr size_t OFF_W3T  = OFF_W2T + 256 * 256 * 2;   // half [64][256]
constexpr size_t OFF_AM1  = OFF_W3T + 64 * 256 * 2;    // half [64][64] x6
constexpr size_t OFF_AM2  = OFF_AM1 + 64 * 64 * 2;
constexpr size_t OFF_AM3  = OFF_AM2 + 64 * 64 * 2;
constexpr size_t OFF_AS1  = OFF_AM3 + 64 * 64 * 2;
constexpr size_t OFF_AS2  = OFF_AS1 + 64 * 64 * 2;
constexpr size_t OFF_AS3  = OFF_AS2 + 64 * 64 * 2;
constexpr size_t OFF_B1   = OFF_AS3 + 64 * 64 * 2;     // f32[256]
constexpr size_t OFF_B2   = OFF_B1 + 256 * 4;          // f32[256]
constexpr size_t OFF_B3   = OFF_B2 + 256 * 4;          // f32[64]
constexpr size_t OFF_AMB1 = OFF_B3 + 64 * 4;
constexpr size_t OFF_AMB2 = OFF_AMB1 + 64 * 4;
constexpr size_t OFF_AMB3 = OFF_AMB2 + 64 * 4;
constexpr size_t OFF_ASB1 = OFF_AMB3 + 64 * 4;
constexpr size_t OFF_ASB2 = OFF_ASB1 + 64 * 4;
constexpr size_t OFF_ASB3 = OFF_ASB2 + 64 * 4;
constexpr size_t OFF_YH   = OFF_ASB3 + 64 * 4;         // half [64][64]  (y, f16)
constexpr size_t OFF_H1   = OFF_YH + 64 * 64 * 2;      // half [64][256] (ODE hidden)
constexpr size_t OFF_HA   = OFF_H1 + 64 * 256 * 2;     // half [64][64]  (actor hidden)
constexpr size_t SMEM_BYTES = OFF_HA + 64 * 64 * 2;    // = 298752 B

// ------------------------------- math --------------------------------------
DEV float tanh_fast(float x) {            // 1 - 2/(e^{2x}+1); saturates correctly
    float e = __expf(2.0f * x);
    return 1.0f - 2.0f / (e + 1.0f);
}
DEV float softplus_f(float x) {           // stable log(1+e^x)
    return fmaxf(x, 0.0f) + log1pf(__expf(-fabsf(x)));
}
DEV float cleanf(float x) { return __builtin_isfinite(x) ? x : 0.0f; }

DEV v8f wmma_f16(v16h a, v16h b, v8f c) {
    return __builtin_amdgcn_wmma_f32_16x16x32_f16(false, a, false, b,
                                                  (short)0, c, false, false);
}

// A fragment: 16x32 f16. lane(row = lane&15, g = lane>>4) holds
// K = k0 + g*16 + {0..15}  -> two 16B LDS loads.
DEV v16h load_a(const _Float16* base, int ld, int m0, int k0, int row, int g) {
    const _Float16* p = base + (m0 + row) * ld + k0 + g * 16;
    v8h lo = *(const v8h*)(p);
    v8h hi = *(const v8h*)(p + 8);
    v16h a;
#pragma unroll
    for (int j = 0; j < 8; ++j) { a[j] = lo[j]; a[j + 8] = hi[j]; }
    return a;
}

// B fragment: 32x16 f16 from W^T stored [N][K]. lane(col = lane&15, g) holds
// K = k0 + g*16 + {0..15} of column n0+col -> 32 contiguous bytes.
DEV v16h load_b(const _Float16* wt, int ld, int n0, int k0, int col, int g) {
    return *(const v16h*)(wt + (n0 + col) * ld + k0 + g * 16);
}

// ---------------------- cooperative weight staging --------------------------
template <int K, int N>
DEV void load_wt(_Float16* dst, const float* __restrict__ src, int tid) {
    // dst[n*K + k] = src[k*N + n]  (global reads coalesced; once per WG)
#pragma unroll 1
    for (int i = tid; i < K * N; i += NTHREADS) {
        int k = i / N, n = i % N;
        dst[n * K + k] = (_Float16)src[i];
    }
}
DEV void copyf(float* dst, const float* __restrict__ src, int n, int tid) {
    for (int i = tid; i < n; i += NTHREADS) dst[i] = src[i];
}

// -------------------- GEMM + activation -> LDS (f16) ------------------------
// C[64][NT*16] = act(A[64][K] @ W[K][NT*16] + bias). 4*NT tiles, tile id
// = wave + 8*it. compute -> barrier -> store -> barrier makes dst==A safe.
enum { ACT_TANH = 0, ACT_SOFTPLUS = 1, ACT_RELU = 2 };

template <int KDIM, int NT, int ACT>
DEV void gemm_act_store(const _Float16* __restrict__ A,
                        const _Float16* __restrict__ WT,
                        const float* __restrict__ bias,
                        _Float16* __restrict__ dst,
                        int wave, int row, int g, int col) {
    constexpr int TPW = NT / 2;
    v8f acc[TPW];
#pragma unroll
    for (int it = 0; it < TPW; ++it) {
        int tile = wave + 8 * it;
        int tm = tile & 3, tn = tile >> 2;
        float bv = bias[tn * 16 + col];
        v8f c;
#pragma unroll
        for (int j = 0; j < 8; ++j) c[j] = bv;
#pragma unroll
        for (int k0 = 0; k0 < KDIM; k0 += 32) {
            v16h a = load_a(A, KDIM, tm * 16, k0, row, g);
            v16h b = load_b(WT, KDIM, tn * 16, k0, col, g);
            c = wmma_f16(a, b, c);
        }
        acc[it] = c;
    }
    __syncthreads();
#pragma unroll
    for (int it = 0; it < TPW; ++it) {
        int tile = wave + 8 * it;
        int tm = tile & 3, tn = tile >> 2;
#pragma unroll
        for (int j = 0; j < 8; ++j) {
            float x = acc[it][j];
            if (ACT == ACT_TANH)          x = tanh_fast(x);
            else if (ACT == ACT_SOFTPLUS) x = softplus_f(x);
            else                          x = fmaxf(x, 0.0f);
            dst[(tm * 16 + j + 8 * g) * (NT * 16) + tn * 16 + col] = (_Float16)x;
        }
    }
    __syncthreads();
}

// ------------- final-layer GEMM (N=64) kept in registers ---------------------
// Tiles 2*wave+{0,1} -> element-aligned with the wave's y / eps / out tiles.
template <int KDIM>
DEV void gemm2reg(const _Float16* __restrict__ A,
                  const _Float16* __restrict__ WT,
                  const float* __restrict__ bias,
                  int wave, int row, int g, int col, v8f& c0, v8f& c1) {
#pragma unroll
    for (int it = 0; it < 2; ++it) {
        int tile = 2 * wave + it;
        int tm = tile & 3, tn = tile >> 2;
        float bv = bias[tn * 16 + col];
        v8f c;
#pragma unroll
        for (int j = 0; j < 8; ++j) c[j] = bv;
#pragma unroll
        for (int k0 = 0; k0 < KDIM; k0 += 32) {
            v16h a = load_a(A, KDIM, tm * 16, k0, row, g);
            v16h b = load_b(WT, KDIM, tn * 16, k0, col, g);
            c = wmma_f16(a, b, c);
        }
        if (it == 0) c0 = c; else c1 = c;
    }
}

// --------------------------------- kernel -----------------------------------
__global__ __launch_bounds__(NTHREADS) void node_sac_fused(
    const float* __restrict__ y0, const float* __restrict__ tvec,
    const float* __restrict__ noise,
    const float* __restrict__ oW1, const float* __restrict__ ob1,
    const float* __restrict__ oW2, const float* __restrict__ ob2,
    const float* __restrict__ oW3, const float* __restrict__ ob3,
    const float* __restrict__ amW1, const float* __restrict__ amb1,
    const float* __restrict__ amW2, const float* __restrict__ amb2,
    const float* __restrict__ amW3, const float* __restrict__ amb3,
    const float* __restrict__ asW1, const float* __restrict__ asb1,
    const float* __restrict__ asW2, const float* __restrict__ asb2,
    const float* __restrict__ asW3, const float* __restrict__ asb3,
    float* __restrict__ out) {
    extern __shared__ __align__(128) char smem[];
    _Float16* W1T = (_Float16*)(smem + OFF_W1T);
    _Float16* W2T = (_Float16*)(smem + OFF_W2T);
    _Float16* W3T = (_Float16*)(smem + OFF_W3T);
    _Float16* AM1 = (_Float16*)(smem + OFF_AM1);
    _Float16* AM2 = (_Float16*)(smem + OFF_AM2);
    _Float16* AM3 = (_Float16*)(smem + OFF_AM3);
    _Float16* AS1 = (_Float16*)(smem + OFF_AS1);
    _Float16* AS2 = (_Float16*)(smem + OFF_AS2);
    _Float16* AS3 = (_Float16*)(smem + OFF_AS3);
    float* B1   = (float*)(smem + OFF_B1);
    float* B2   = (float*)(smem + OFF_B2);
    float* B3   = (float*)(smem + OFF_B3);
    float* AMB1 = (float*)(smem + OFF_AMB1);
    float* AMB2 = (float*)(smem + OFF_AMB2);
    float* AMB3 = (float*)(smem + OFF_AMB3);
    float* ASB1 = (float*)(smem + OFF_ASB1);
    float* ASB2 = (float*)(smem + OFF_ASB2);
    float* ASB3 = (float*)(smem + OFF_ASB3);
    _Float16* Yh = (_Float16*)(smem + OFF_YH);
    _Float16* H1 = (_Float16*)(smem + OFF_H1);
    _Float16* HA = (_Float16*)(smem + OFF_HA);

    const int tid  = threadIdx.x;
    const int wave = tid >> 5;
    const int lane = tid & 31;
    const int row  = lane & 15;   // M/N index within 16
    const int col  = row;
    const int g    = lane >> 4;   // lane half-group

    // ---- stage all weights into LDS (fp16, transposed), once per WG ----
    load_wt<64, 256>(W1T, oW1, tid);
    load_wt<256, 256>(W2T, oW2, tid);
    load_wt<256, 64>(W3T, oW3, tid);
    load_wt<64, 64>(AM1, amW1, tid);
    load_wt<64, 64>(AM2, amW2, tid);
    load_wt<64, 64>(AM3, amW3, tid);
    load_wt<64, 64>(AS1, asW1, tid);
    load_wt<64, 64>(AS2, asW2, tid);
    load_wt<64, 64>(AS3, asW3, tid);
    copyf(B1, ob1, 256, tid);
    copyf(B2, ob2, 256, tid);
    copyf(B3, ob3, 64, tid);
    copyf(AMB1, amb1, 64, tid);
    copyf(AMB2, amb2, 64, tid);
    copyf(AMB3, amb3, 64, tid);
    copyf(ASB1, asb1, 64, tid);
    copyf(ASB2, asb2, 64, tid);
    copyf(ASB3, asb3, 64, tid);

    const int rb = blockIdx.x * ROWS_PER_WG;  // global row base
    const float dt = tvec[1] - tvec[0];

    // ---- load y0 into per-lane f32 registers (C-fragment layout) ----
    float yv[2][8];
#pragma unroll
    for (int it = 0; it < 2; ++it) {
        int tile = 2 * wave + it;
        int tm = tile & 3, tn = tile >> 2;
#pragma unroll
        for (int j = 0; j < 8; ++j) {
            int lr = tm * 16 + j + 8 * g;
            int lc = tn * 16 + col;
            float v = y0[(size_t)(rb + lr) * Dq + lc];
            yv[it][j] = v;
            out[(size_t)(rb + lr) * Dq + lc] = v;      // out[0] = y0
            Yh[lr * Dq + lc] = (_Float16)v;
        }
    }
    __syncthreads();

    // --------------------------- Euler time loop ---------------------------
#pragma unroll 1
    for (int step = 0; step < Tq - 1; ++step) {
        // ODE func: h1 = tanh(y@W1+b1); h2 = softplus(h1@W2+b2); f = h2@W3+b3
        gemm_act_store<64, 16, ACT_TANH>(Yh, W1T, B1, H1, wave, row, g, col);
        gemm_act_store<256, 16, ACT_SOFTPLUS>(H1, W2T, B2, H1, wave, row, g, col);
        v8f f0, f1;
        gemm2reg<256>(H1, W3T, B3, wave, row, g, col, f0, f1);

        // actor mean: relu, relu, tanh
        gemm_act_store<64, 4, ACT_RELU>(Yh, AM1, AMB1, HA, wave, row, g, col);
        gemm_act_store<64, 4, ACT_RELU>(HA, AM2, AMB2, HA, wave, row, g, col);
        v8f m0, m1;
        gemm2reg<64>(HA, AM3, AMB3, wave, row, g, col, m0, m1);

        // actor log_std: relu, relu, linear (barrier inside stage guards HA reuse)
        gemm_act_store<64, 4, ACT_RELU>(Yh, AS1, ASB1, HA, wave, row, g, col);
        gemm_act_store<64, 4, ACT_RELU>(HA, AS2, ASB2, HA, wave, row, g, col);
        v8f s0, s1;
        gemm2reg<64>(HA, AS3, ASB3, wave, row, g, col, s0, s1);

        const float* eps = noise + (size_t)step * Bq * Dq;
        float* outp = out + (size_t)(step + 1) * Bq * Dq;

#pragma unroll
        for (int it = 0; it < 2; ++it) {
            int tile = 2 * wave + it;
            int tm = tile & 3, tn = tile >> 2;
            v8f fv = it ? f1 : f0;
            v8f mv = it ? m1 : m0;
            v8f sv = it ? s1 : s0;
#pragma unroll
            for (int j = 0; j < 8; ++j) {
                int lr = tm * 16 + j + 8 * g;
                int lc = tn * 16 + col;
                float mean = cleanf(tanh_fast(mv[j]));
                float lstd = cleanf(sv[j]);
                float stdv = softplus_f(lstd);
                float e = eps[(size_t)(rb + lr) * Dq + lc];
                float u = tanh_fast(mean + stdv * e);   // tanh(rsample)
                float yn = yv[it][j] + (fv[j] - u) * dt;
                yv[it][j] = yn;
                outp[(size_t)(rb + lr) * Dq + lc] = yn;
                Yh[lr * Dq + lc] = (_Float16)yn;
            }
        }
        __syncthreads();  // Yh ready for next step's GEMMs
    }
}

// ------------------------------ host launcher -------------------------------
extern "C" void kernel_launch(void* const* d_in, const int* in_sizes, int n_in,
                              void* d_out, int out_size, void* d_ws, size_t ws_size,
                              hipStream_t stream) {
    (void)in_sizes; (void)n_in; (void)out_size; (void)d_ws; (void)ws_size;
    const float* y0    = (const float*)d_in[0];
    const float* tvec  = (const float*)d_in[1];
    const float* noise = (const float*)d_in[2];
    const float* oW1 = (const float*)d_in[3];  const float* ob1 = (const float*)d_in[4];
    const float* oW2 = (const float*)d_in[5];  const float* ob2 = (const float*)d_in[6];
    const float* oW3 = (const float*)d_in[7];  const float* ob3 = (const float*)d_in[8];
    const float* amW1 = (const float*)d_in[9];  const float* amb1 = (const float*)d_in[10];
    const float* amW2 = (const float*)d_in[11]; const float* amb2 = (const float*)d_in[12];
    const float* amW3 = (const float*)d_in[13]; const float* amb3 = (const float*)d_in[14];
    const float* asW1 = (const float*)d_in[15]; const float* asb1 = (const float*)d_in[16];
    const float* asW2 = (const float*)d_in[17]; const float* asb2 = (const float*)d_in[18];
    const float* asW3 = (const float*)d_in[19]; const float* asb3 = (const float*)d_in[20];
    float* out = (float*)d_out;

    // Allow >64KB dynamic LDS (idempotent; safe under graph capture).
    (void)hipFuncSetAttribute((const void*)node_sac_fused,
                              hipFuncAttributeMaxDynamicSharedMemorySize,
                              (int)SMEM_BYTES);

    dim3 grid(Bq / ROWS_PER_WG);   // 256 workgroups
    dim3 block(NTHREADS);          // 8 wave32
    node_sac_fused<<<grid, block, SMEM_BYTES, stream>>>(
        y0, tvec, noise,
        oW1, ob1, oW2, ob2, oW3, ob3,
        amW1, amb1, amW2, amb2, amW3, amb3,
        asW1, asb1, asW2, asb2, asW3, asb3,
        out);
}
